// Spectrogram_32031866093829
// MI455X (gfx1250) — matrix operand order repaired and checked by reading, not compiled, not run on hardware
//
#include <hip/hip_runtime.h>

typedef __attribute__((ext_vector_type(16))) _Float16 v16h;
typedef __attribute__((ext_vector_type(8)))  _Float16 v8h;
typedef __attribute__((ext_vector_type(8)))  float    v8f;

namespace {
constexpr int kNfft    = 512;
constexpr int kHop     = 160;
constexpr int kNbin    = 257;
constexpr int kBatch   = 32;
constexpr int kSamples = 320000;
constexpr int kTout    = 2001;   // (320000)/160 + 1 with 256-reflect pad, K=512
constexpr int kPad     = 256;

constexpr int kTPerBlock = 128;  // 8 waves * 16 rows
constexpr int kThreads   = 256;
constexpr int kSpan      = kHop * (kTPerBlock - 1) + kNfft;  // 20832 samples staged
constexpr int kApad      = 8;    // pad halves inserted per 160-half hop (bank spread)
constexpr int kARegion   = 21888;                 // >= 20831 + 130*8 + 1, 16B-aligned
constexpr int kWStride   = 520;                   // 512 + 8 halves (bank spread)
constexpr int kWRegion   = 4 * 16 * kWStride;     // [r_hi, r_lo, i_hi, i_lo][16 bins]
constexpr int kSmemHalves = 2 * kARegion + kWRegion;  // 77056 halves = 154112 B
}

__device__ __forceinline__ v16h cat8(v8h x, v8h y) {
  return __builtin_shufflevector(x, y, 0,1,2,3,4,5,6,7,8,9,10,11,12,13,14,15);
}

__global__ __launch_bounds__(kThreads)
void spectrogram_wmma_kernel(const float* __restrict__ wf,
                             const float* __restrict__ wr,
                             const float* __restrict__ wi,
                             float* __restrict__ out) {
  extern __shared__ _Float16 sm[];
  _Float16* a_hi = sm;                       // padded frame span, hi halves
  _Float16* a_lo = sm + kARegion;            // padded frame span, lo halves
  _Float16* wlds = sm + 2 * kARegion;        // [kind][bin][kWStride]

  const int tb  = blockIdx.x * kTPerBlock;   // first output frame of block
  const int nb  = blockIdx.y * 16;           // first frequency bin of block
  const int b   = blockIdx.z;
  const int tid = threadIdx.x;

  // ---- stage reflect-padded frame span into LDS as f16 hi/lo split ----
  const long sbase = (long)b * kSamples;
  const int  p0g   = tb * kHop;              // padded-signal index of span start
  for (int i = tid; i < kSpan; i += kThreads) {
    int s = p0g + i - kPad;                      // reflect pad: |s|, tail mirror
    s = (s < 0) ? -s : s;
    s = (s >= kSamples) ? (2 * kSamples - 2 - s) : s;
    float x = wf[sbase + s];
    _Float16 h = (_Float16)x;
    _Float16 l = (_Float16)(x - (float)h);
    int p = i + (i / kHop) * kApad;              // bank-spread padded address
    a_hi[p] = h;
    a_lo[p] = l;
  }

  // ---- stage 16-bin weight tile (real/imag, hi/lo) into LDS ----
  for (int i = tid; i < 16 * kNfft; i += kThreads) {
    int bin = i >> 9;
    int k   = i & (kNfft - 1);
    int row = nb + bin;
    float xr = 0.0f, xi = 0.0f;
    if (row < kNbin) {
      xr = wr[row * kNfft + k];
      xi = wi[row * kNfft + k];
    }
    _Float16 rh = (_Float16)xr; _Float16 rl = (_Float16)(xr - (float)rh);
    _Float16 ih = (_Float16)xi; _Float16 il = (_Float16)(xi - (float)ih);
    int base = bin * kWStride + k;
    wlds[0 * 16 * kWStride + base] = rh;
    wlds[1 * 16 * kWStride + base] = rl;
    wlds[2 * 16 * kWStride + base] = ih;
    wlds[3 * 16 * kWStride + base] = il;
  }
  __syncthreads();

  const int wv   = tid >> 5;        // wave id: t-tile within block
  const int lane = tid & 31;
  const int lh   = lane & 15;       // A row (frame) / B col (bin) / D col
  const int hb   = lane >> 4;       // half-wave select

  v8f accr = {};
  v8f acci = {};

  // A row base in padded LDS span: row stride = (160 + 8) halves = 336 B.
  // Fold in the half-wave K offset (16*hb) so the unrolled loop only needs
  // compile-time constant ds offsets.
  const _Float16* aHi = a_hi + (16 * wv + lh) * (kHop + kApad) + 16 * hb;
  const _Float16* aLo = a_lo + (16 * wv + lh) * (kHop + kApad) + 16 * hb;
  // B fragment base: 16 consecutive halves at kc + 16*hb, row = bin lh
  const _Float16* wB  = wlds + lh * kWStride + 16 * hb;

  #pragma unroll
  for (int kc = 0; kc < kNfft; kc += 32) {
    // Segment index of the pad-inserting A layout is hb-independent and a
    // compile-time constant under full unroll (kc multiple of 32, hops of 160).
    const int o0 = kc + (kc / kHop) * kApad;              // run at K = kc + 16*hb
    const int o1 = kc + 8 + ((kc + 8) / kHop) * kApad;  // run at K = kc+8+16*hb
    v16h Ah = cat8(*(const v8h*)(aHi + o0), *(const v8h*)(aHi + o1));
    v16h Al = cat8(*(const v8h*)(aLo + o0), *(const v8h*)(aLo + o1));

    const _Float16* wp = wB + kc;
    v16h Brh = cat8(*(const v8h*)(wp + 0 * 16 * kWStride),
                    *(const v8h*)(wp + 0 * 16 * kWStride + 8));
    v16h Brl = cat8(*(const v8h*)(wp + 1 * 16 * kWStride),
                    *(const v8h*)(wp + 1 * 16 * kWStride + 8));
    v16h Bih = cat8(*(const v8h*)(wp + 2 * 16 * kWStride),
                    *(const v8h*)(wp + 2 * 16 * kWStride + 8));
    v16h Bil = cat8(*(const v8h*)(wp + 3 * 16 * kWStride),
                    *(const v8h*)(wp + 3 * 16 * kWStride + 8));

    // fp32-emulating f16 split GEMM: Ah*Bh + Ah*Bl + Al*Bh (fp32 accumulate),
    // real/imag interleaved to relax accumulator dependency chains.
    accr = __builtin_amdgcn_wmma_f32_16x16x32_f16(false, Ah, false, Brh, (short)0, accr, false, false);
    acci = __builtin_amdgcn_wmma_f32_16x16x32_f16(false, Ah, false, Bih, (short)0, acci, false, false);
    accr = __builtin_amdgcn_wmma_f32_16x16x32_f16(false, Ah, false, Brl, (short)0, accr, false, false);
    acci = __builtin_amdgcn_wmma_f32_16x16x32_f16(false, Ah, false, Bil, (short)0, acci, false, false);
    accr = __builtin_amdgcn_wmma_f32_16x16x32_f16(false, Al, false, Brh, (short)0, accr, false, false);
    acci = __builtin_amdgcn_wmma_f32_16x16x32_f16(false, Al, false, Bih, (short)0, acci, false, false);
  }

  // ---- fused power + store: D layout row M = v + 8*hb, col N = lh ----
  const int bin = nb + lh;
  if (bin < kNbin) {
    const long obase = (long)b * kTout * kNbin;
    #pragma unroll
    for (int v = 0; v < 8; ++v) {
      int t = tb + 16 * wv + v + 8 * hb;
      if (t < kTout) {
        float r = accr[v];
        float im = acci[v];
        out[obase + (long)t * kNbin + bin] = r * r + im * im;
      }
    }
  }
}

extern "C" void kernel_launch(void* const* d_in, const int* in_sizes, int n_in,
                              void* d_out, int out_size, void* d_ws, size_t ws_size,
                              hipStream_t stream) {
  (void)in_sizes; (void)n_in; (void)out_size; (void)d_ws; (void)ws_size;
  const float* wf = (const float*)d_in[0];   // (32, 320000) f32
  const float* wr = (const float*)d_in[1];   // (257, 1, 512) f32
  const float* wi = (const float*)d_in[2];   // (257, 1, 512) f32
  float* out = (float*)d_out;                // (32, 1, 2001, 257) f32

  dim3 grid((kTout + kTPerBlock - 1) / kTPerBlock,   // 16 t-groups
            (kNbin + 15) / 16,                       // 17 bin tiles
            kBatch);                                 // 32
  size_t shmem = (size_t)kSmemHalves * 2;            // 154112 bytes
  spectrogram_wmma_kernel<<<grid, kThreads, shmem, stream>>>(wf, wr, wi, out);
}
